// AttnEncoder_44762149159349
// MI455X (gfx1250) — matrix operand order repaired and checked by reading, not compiled, not run on hardware
//
#include <hip/hip_runtime.h>
#include <hip/hip_bf16.h>
#include <math.h>

typedef __bf16 bf16;
typedef __attribute__((ext_vector_type(16))) __bf16 v16bf;
typedef __attribute__((ext_vector_type(8)))  __bf16 v8bf;
typedef __attribute__((ext_vector_type(8)))  float  v8f;
typedef int v4i __attribute__((ext_vector_type(4)));

#define S  256
#define B  64
#define H  512
#define IND 1024
#define SB (S*B)        // 16384
#define NWG 8           // cooperating workgroups per direction in recurrence

// workspace layout (bytes)
#define OFF_BAR   0ull
#define OFF_H     1024ull
#define OFF_RH    (OFF_H  + 2ull*B*H*2ull)            // 132096
#define OFF_X     (OFF_RH + 2ull*B*H*2ull)            // 263168
#define OFF_OUT0  (OFF_X  + (unsigned long long)SB*H*2ull)
#define OFF_WP    (OFF_OUT0 + 2ull*SB*H*2ull)
#define OFF_A     (OFF_WP + 24ull*H*H*2ull)
#define WS_NEED   (OFF_A  + 6ull*SB*H*2ull)

#define RECUR_LDS (196608 + 65536 + 16384 + 8192)     // 286720 B  (< 320 KB)

// ---------------- async global -> LDS staging (CDNA5 ASYNCcnt path) ---------
#if __has_builtin(__builtin_amdgcn_global_load_async_to_lds_b128)
#define USE_ASYNC 1
typedef __attribute__((address_space(1))) v4i* g_v4i_ptr;   // global int4*
typedef __attribute__((address_space(3))) v4i* l_v4i_ptr;   // LDS int4*
__device__ __forceinline__ void g2l_b128(void* l, const void* g) {
  __builtin_amdgcn_global_load_async_to_lds_b128((g_v4i_ptr)g, (l_v4i_ptr)l, 0, 0);
}
__device__ __forceinline__ void async_wait() {
  asm volatile("s_wait_asynccnt 0" ::: "memory");
}
#else
#define USE_ASYNC 0
__device__ __forceinline__ void async_wait() {}
#endif

// cooperative copy of n_v8 * 16B from global to LDS
__device__ __forceinline__ void copy_g2l(bf16* ldst, const bf16* gsrc, int n_v8,
                                         int tid, int nth) {
#if USE_ASYNC
  for (int it = tid; it < n_v8; it += nth)
    g2l_b128((void*)(ldst + it * 8), (const void*)(gsrc + it * 8));
#else
  for (int it = tid; it < n_v8; it += nth)
    ((v8bf*)ldst)[it] = ((const v8bf*)gsrc)[it];
#endif
}
// ----------------------------------------------------------------------------

__device__ __forceinline__ v8f wmma_bf16(v16bf a, v16bf b, v8f c) {
  return __builtin_amdgcn_wmma_f32_16x16x32_bf16(false, a, false, b, (short)0, c,
                                                 false, false);
}

// A-operand fragment: 16x32 bf16 tile from a row-major [.,ld] buffer.
// lane<16: row=mt*16+lane,  K = kt*32 + {0..7, 16..23}
// lane>=16: row=mt*16+lane-16, K = kt*32 + {8..15, 24..31}
__device__ __forceinline__ v16bf load_a_frag(const bf16* base, int ld, int mt,
                                             int kt, int lane) {
  int r  = mt * 16 + (lane & 15);
  int k0 = kt * 32 + ((lane >> 4) << 3);
  const bf16* p = base + r * ld + k0;
  v8bf lo = *(const v8bf*)p;
  v8bf hi = *(const v8bf*)(p + 16);
  return __builtin_shufflevector(lo, hi, 0,1,2,3,4,5,6,7,8,9,10,11,12,13,14,15);
}

// inter-workgroup sense barrier (NWG workgroups per direction)
__device__ __forceinline__ void ws_barrier(unsigned* bar, unsigned nwg) {
  __threadfence();
  __syncthreads();
  if (threadIdx.x == 0) {
    unsigned g = __hip_atomic_load(&bar[1], __ATOMIC_ACQUIRE, __HIP_MEMORY_SCOPE_AGENT);
    unsigned v = __hip_atomic_fetch_add(&bar[0], 1u, __ATOMIC_ACQ_REL, __HIP_MEMORY_SCOPE_AGENT);
    if (v == nwg - 1u) {
      __hip_atomic_store(&bar[0], 0u, __ATOMIC_RELAXED, __HIP_MEMORY_SCOPE_AGENT);
      __hip_atomic_fetch_add(&bar[1], 1u, __ATOMIC_RELEASE, __HIP_MEMORY_SCOPE_AGENT);
    } else {
      while (__hip_atomic_load(&bar[1], __ATOMIC_ACQUIRE, __HIP_MEMORY_SCOPE_AGENT) == g)
        __builtin_amdgcn_s_sleep(2);
    }
  }
  __syncthreads();
}

__global__ void k_init(unsigned* w, int n) {
  int i = blockIdx.x * blockDim.x + threadIdx.x;
  if (i < n) w[i] = 0u;
}

__global__ void k_gather(const int* tok, const float* emb, bf16* x) {
  int sb = blockIdx.x;                 // (s*64 + b)
  long row = (long)tok[sb] * H;
  bf16* dst = x + (long)sb * H;
  for (int i = threadIdx.x; i < H; i += blockDim.x)
    dst[i] = (bf16)emb[row + i];
}

// pack W into per-tile B-operand lane layout:
// wp[p][nt(32)][kt(16)][lane(32)][i(16)]   (p = ((d*2+layer)*3+g)*2+part)
// element = W[d,layer][n = nt*16 + (lane&15)][k = part*512 + kt*32 + (lane>=16?8:0) + (i&7) + (i>=8?16:0)]
__global__ void k_pack(const float* Wr, const float* Wz, const float* Ws, bf16* wp) {
  int e = blockIdx.x * blockDim.x + threadIdx.x;   // 24 * 262144 total
  int p    = e >> 18;
  int q    = e & 262143;
  int part = p & 1;
  int t1   = p >> 1;
  int g    = t1 % 3;
  int t2   = t1 / 3;
  int layer = t2 & 1;
  int d     = t2 >> 1;
  int nt = q >> 13;  int q2 = q & 8191;
  int kt = q2 >> 9;  int q3 = q2 & 511;
  int lane = q3 >> 4; int i = q3 & 15;
  int n = nt * 16 + (lane & 15);
  int k = part * 512 + kt * 32 + ((lane >> 4) << 3) + (i & 7) + ((i >> 3) << 4);
  const float* W = (g == 0) ? Wr : (g == 1) ? Wz : Ws;
  wp[e] = (bf16)W[((long)(d * 2 + layer) * H + n) * IND + k];
}

// Input-half GEMM: A[d][g][t][b][n] = in_row(t or flipped) @ Wx.T + bias
__global__ void k_agemm(const bf16* in, long dirStride, int flip,
                        const bf16* wp, const float* br, const float* bz,
                        const float* bs, bf16* A, int layer) {
  extern __shared__ char smem[];
  bf16* tile = (bf16*)smem;            // 64 x 512 bf16 = 64 KB
  int t   = blockIdx.x >> 3;
  int n64 = blockIdx.x & 7;
  int dg  = blockIdx.y;
  int d = dg / 3, g = dg % 3;
  int tsrc = (flip && d == 1) ? (S - 1 - t) : t;
  const bf16* src = in + d * dirStride + (long)tsrc * B * H;
  copy_g2l(tile, src, (B * H) / 8, threadIdx.x, blockDim.x);
  async_wait();
  __syncthreads();

  int lane = threadIdx.x & 31, wave = threadIdx.x >> 5;
  const bf16* wpg = wp + (((long)((d * 2 + layer) * 3 + g) * 2 + 0) << 18);
  const float* bias = (g == 0) ? br : (g == 1) ? bz : bs;

  for (int tt = wave; tt < 16; tt += 8) {
    int mt = tt >> 2, ntl = tt & 3, nt = n64 * 4 + ntl;
    v8f acc = {};
    for (int kt = 0; kt < 16; ++kt) {
      v16bf a = load_a_frag(tile, H, mt, kt, lane);
      v16bf b = *(const v16bf*)(wpg + ((nt * 16 + kt) << 9) + lane * 16);
      acc = wmma_bf16(a, b, acc);
    }
    int nglob = nt * 16 + (lane & 15);
    float bv = bias[(d * 2 + layer) * H + nglob];
    int mhi = ((lane >> 4) << 3) + mt * 16;
    for (int v = 0; v < 8; ++v) {
      int m = mhi + v;
      A[((long)(d * 3 + g) * SB + (long)t * B + m) * H + nglob] = (bf16)(acc[v] + bv);
    }
  }
}

// Cooperative recurrence: 2 dirs x NWG workgroups; each WG owns 64 hidden cols,
// keeps its Wh slices (3 gates, 192 KB) resident in LDS for all 256 steps.
__global__ void k_recur(const bf16* A, const bf16* wp, bf16* hG, bf16* rhG,
                        bf16* out0, float* dout, unsigned* bar, int layer) {
  extern __shared__ char smem[];
  bf16*  wlds = (bf16*)smem;                                  // 3*4*16*512
  bf16*  hlds = (bf16*)(smem + 196608);                       // 64*512
  float* zlds = (float*)(smem + 196608 + 65536);              // 64*64
  bf16*  hsl  = (bf16*)(smem + 196608 + 65536 + 16384);       // 64*64
  int d = blockIdx.x / NWG, wg = blockIdx.x % NWG;
  int lane = threadIdx.x & 31, wave = threadIdx.x >> 5;
  unsigned* mybar = bar + d * 8;

  // stage h-part weight slices into LDS once (async, VGPR-free)
  for (int c = 0; c < 12; ++c) {
    int g = c / 4, ntl = c % 4, nt = wg * 4 + ntl;
    const bf16* src = wp + (((long)((d * 2 + layer) * 3 + g) * 2 + 1) << 18)
                         + ((long)(nt * 16) << 9);
    bf16* dst = wlds + ((long)((g * 4 + ntl) * 16) << 9);
    copy_g2l(dst, src, (16 * 512) / 8, threadIdx.x, blockDim.x);
  }
  async_wait();
  bf16* hGd  = hG  + d * B * H;
  bf16* rhGd = rhG + d * B * H;
  ws_barrier(mybar, NWG);   // weights staged, h zeroed by init kernel

  for (int t = 0; t < S; ++t) {
    // full h_t -> LDS
    copy_g2l(hlds, hGd, (B * H) / 8, threadIdx.x, blockDim.x);
    async_wait();
    __syncthreads();

    // r and z gates on this WG's 64 columns
    for (int g = 0; g < 2; ++g) {
      for (int tt = wave; tt < 16; tt += 8) {
        int mt = tt >> 2, ntl = tt & 3;
        int nloc = ntl * 16 + (lane & 15);
        int nglob = wg * 64 + nloc;
        int mhi = ((lane >> 4) << 3) + mt * 16;
        const bf16* Ap = A + ((long)(d * 3 + g) * SB + (long)t * B) * H;
        v8f acc;
        for (int v = 0; v < 8; ++v) acc[v] = (float)Ap[(mhi + v) * H + nglob];
        for (int kt = 0; kt < 16; ++kt) {
          v16bf a = load_a_frag(hlds, H, mt, kt, lane);
          v16bf b = *(const v16bf*)(wlds + (((g * 4 + ntl) * 16 + kt) << 9) + lane * 16);
          acc = wmma_bf16(a, b, acc);
        }
        for (int v = 0; v < 8; ++v) {
          int m = mhi + v;
          float gate = 1.0f / (1.0f + __expf(-acc[v]));
          if (g == 0) {
            float hv = (float)hlds[m * H + nglob];
            rhGd[m * H + nglob] = (bf16)(gate * hv);
          } else {
            zlds[m * 64 + nloc] = gate;
            hsl[m * 64 + nloc]  = hlds[m * H + nglob];
          }
        }
      }
    }
    ws_barrier(mybar, NWG);

    // full (r*h) -> LDS (reuse h buffer)
    copy_g2l(hlds, rhGd, (B * H) / 8, threadIdx.x, blockDim.x);
    async_wait();
    __syncthreads();

    // s gate + state update
    for (int tt = wave; tt < 16; tt += 8) {
      int mt = tt >> 2, ntl = tt & 3;
      int nloc = ntl * 16 + (lane & 15);
      int nglob = wg * 64 + nloc;
      int mhi = ((lane >> 4) << 3) + mt * 16;
      const bf16* Ap = A + ((long)(d * 3 + 2) * SB + (long)t * B) * H;
      v8f acc;
      for (int v = 0; v < 8; ++v) acc[v] = (float)Ap[(mhi + v) * H + nglob];
      for (int kt = 0; kt < 16; ++kt) {
        v16bf a = load_a_frag(hlds, H, mt, kt, lane);
        v16bf b = *(const v16bf*)(wlds + (((2 * 4 + ntl) * 16 + kt) << 9) + lane * 16);
        acc = wmma_bf16(a, b, acc);
      }
      for (int v = 0; v < 8; ++v) {
        int m = mhi + v;
        float sv = tanhf(acc[v]);
        float z  = zlds[m * 64 + nloc];
        float hv = (float)hsl[m * 64 + nloc];
        float hn = z * hv + (1.0f - z) * sv;
        hGd[m * H + nglob] = (bf16)hn;
        if (layer == 0)
          out0[((long)d * SB + (long)t * B + m) * H + nglob] = (bf16)hn;
        else
          dout[((long)t * B + m) * (2 * H) + d * H + nglob] = hn;
        if (t == S - 1)
          dout[(long)S * B * 2 * H + ((long)(d * 2 + layer) * B + m) * H + nglob] = hn;
      }
    }
    ws_barrier(mybar, NWG);
  }
}

extern "C" void kernel_launch(void* const* d_in, const int* in_sizes, int n_in,
                              void* d_out, int out_size, void* d_ws, size_t ws_size,
                              hipStream_t stream) {
  (void)in_sizes; (void)n_in; (void)out_size;
  const int*   tok = (const int*)d_in[0];
  const float* emb = (const float*)d_in[1];
  const float* Wr  = (const float*)d_in[2];
  const float* Wz  = (const float*)d_in[3];
  const float* Ws  = (const float*)d_in[4];
  const float* br  = (const float*)d_in[5];
  const float* bz  = (const float*)d_in[6];
  const float* bs  = (const float*)d_in[7];
  float* out = (float*)d_out;
  char*  ws  = (char*)d_ws;
  if (ws_size < WS_NEED) return;

  unsigned* bar  = (unsigned*)(ws + OFF_BAR);
  bf16* hG   = (bf16*)(ws + OFF_H);
  bf16* rhG  = (bf16*)(ws + OFF_RH);
  bf16* x    = (bf16*)(ws + OFF_X);
  bf16* out0 = (bf16*)(ws + OFF_OUT0);
  bf16* wp   = (bf16*)(ws + OFF_WP);
  bf16* A    = (bf16*)(ws + OFF_A);

  (void)hipFuncSetAttribute((const void*)k_agemm,
                            hipFuncAttributeMaxDynamicSharedMemorySize, 65536);
  (void)hipFuncSetAttribute((const void*)k_recur,
                            hipFuncAttributeMaxDynamicSharedMemorySize, RECUR_LDS);

  // zero barriers + h + rh
  k_init<<<(65792 + 255) / 256, 256, 0, stream>>>((unsigned*)ws, 65792);
  k_gather<<<SB, 256, 0, stream>>>(tok, emb, x);
  k_pack<<<24576, 256, 0, stream>>>(Wr, Wz, Ws, wp);

  // layer 0
  k_agemm<<<dim3(2048, 6), 256, 65536, stream>>>(x, 0L, 1, wp, br, bz, bs, A, 0);
  k_recur<<<2 * NWG, 256, RECUR_LDS, stream>>>(A, wp, hG, rhG, out0, out, bar, 0);

  // reset h0 = 0 for layer 1
  k_init<<<256, 256, 0, stream>>>((unsigned*)(ws + OFF_H), 65536);

  // layer 1
  k_agemm<<<dim3(2048, 6), 256, 65536, stream>>>(out0, (long)SB * H, 0, wp, br, bz, bs, A, 1);
  k_recur<<<2 * NWG, 256, RECUR_LDS, stream>>>(A, wp, hG, rhG, out0, out, bar, 1);
}
